// StructGraphGNN_5471788335203
// MI455X (gfx1250) — matrix operand order repaired and checked by reading, not compiled, not run on hardware
//
#include <hip/hip_runtime.h>
#include <hip/hip_bf16.h>
#include <math.h>

#define NNODES 100000
#define NEDGES 800000
#define NGRAPH 256
#define NH     512
#define NMEMS  8
#define NCLS   10

typedef __attribute__((ext_vector_type(16))) __bf16    v16bf;
typedef __attribute__((ext_vector_type(8)))  float     v8f;
typedef __attribute__((ext_vector_type(4)))  unsigned  v4u;
typedef __attribute__((ext_vector_type(8)))  int       v8i;
typedef __attribute__((ext_vector_type(4)))  int       v4i;

union FragBF { uint4 q[2]; v16bf v; };
union BF8    { uint4 q;    __bf16 h[8]; };

#if defined(__HIP_DEVICE_COMPILE__) && defined(__has_builtin)
#  if __has_builtin(__builtin_amdgcn_tensor_load_to_lds) && \
      __has_builtin(__builtin_amdgcn_s_wait_tensorcnt)
#    define USE_TDM 1
#  else
#    define USE_TDM 0
#  endif
#else
#  define USE_TDM 0
#endif

#if __has_include(<hip/amd_detail/amd_gfx1250_TDM.h>)
#  define TDM_ARITY6 1
#else
#  define TDM_ARITY6 0
#endif

// order-preserving float<->uint encoding so float-max can use integer atomicMax
__device__ __forceinline__ unsigned encOrd(float f) {
  unsigned u = __float_as_uint(f);
  return (u & 0x80000000u) ? ~u : (u | 0x80000000u);
}
__device__ __forceinline__ float decOrd(unsigned u) {
  return (u & 0x80000000u) ? __uint_as_float(u & 0x7FFFFFFFu) : __uint_as_float(~u);
}

#if USE_TDM
// Issue one TDM 2D tile load: [tileY rows x 32 bf16] from a row-major tensor
// (row stride strideElems, elements = bf16) into LDS at byte offset ldsByte.
// Rows beyond tensY read as zero (hardware OOB), data_size = 2 bytes.
__device__ __forceinline__ void tdm2d(const void* g, unsigned ldsByte,
                                      int tileY, long long strideElems, int tensY)
{
  unsigned long long ga = (unsigned long long)g;
  v4u g0 = { 1u,                                   // count=1 (valid user descriptor)
             ldsByte,                              // lds_addr
             (unsigned)ga,                         // global_addr[31:0]
             (unsigned)(((ga >> 32) & 0x1FFFFFFull) | (2u << 30)) }; // addr[56:32] | type=2
  unsigned long long s0 = (unsigned long long)strideElems;
  // D# group1 bit packing (bits relative to 256-bit group):
  //  [17:16] data_size=1 (2B)  [79:48] tensor_dim0  [111:80] tensor_dim1
  //  [127:112] tile_dim0  [143:128] tile_dim1  [207:160] tensor_dim0_stride
  unsigned long long q0 = 0x10000ull | (32ull << 48);                 // dsize=2B, tensor_dim0=32
  unsigned long long q1 = ((unsigned long long)(unsigned)tensY << 16) // tensor_dim1
                        | (32ull << 48);                              // tile_dim0=32
  unsigned long long q2 = (unsigned long long)(unsigned short)tileY   // tile_dim1
                        | ((s0 & 0xFFFFFFFFull) << 32);               // stride0[31:0]
  unsigned long long q3 = (s0 >> 32) & 0xFFFFull;                     // stride0[47:32]
  union { unsigned long long q[4]; v8i v; } g1;
  g1.q[0] = q0; g1.q[1] = q1; g1.q[2] = q2; g1.q[3] = q3;
  v4i z4 = {0, 0, 0, 0};
#if TDM_ARITY6
  v8i z8 = {0, 0, 0, 0, 0, 0, 0, 0};
  __builtin_amdgcn_tensor_load_to_lds(g0, g1.v, z4, z4, z8, 0);
#else
  __builtin_amdgcn_tensor_load_to_lds(g0, g1.v, z4, z4, 0);
#endif
}
#endif

// ---------------------------------------------------------------------------
// GEMM:  C = act( A0@B0 [+ A1@B1] + bias )
// A*: bf16 [M,K] row-major (lda in elems). Bt: bf16 [512][K or 2K] row-major
// pre-transposed weights (ldb in elems); source s uses columns s*K..s*K+K-1.
// C: bf16 [M,512]. bf16 WMMA, f32 accumulate. Workgroup tile 128x64, 8 waves,
// each wave a 32x32 tile. Double-buffered LDS; TDM DMA pipeline when available.
// ---------------------------------------------------------------------------
__global__ __launch_bounds__(256)
void k_gemm_wmma(const __bf16* __restrict__ A0, int lda0,
                 const __bf16* __restrict__ A1, int lda1,
                 const __bf16* __restrict__ Bt, int ldb,
                 const float* __restrict__ bias,
                 __bf16* __restrict__ C, int ldc,
                 int M, int K, int do_relu)
{
  // per buffer: A tile 128x32 (4096 elems) then B tile 64x32 (2048 elems)
  __shared__ alignas(16) __bf16 sAB[2][4096 + 2048];   // 24 KB

  const int tid  = threadIdx.x;
  const int lane = tid & 31;
  const int wave = tid >> 5;
  const int wm   = wave >> 1;                  // 0..3
  const int wn   = wave & 1;                   // 0..1
  const int rowBase = blockIdx.x * 128;
  const int colBase = blockIdx.y * 64;
  const int kSteps  = K >> 5;
  const int steps   = (A1 ? 2 : 1) * kSteps;

  v8f acc[2][2] = {};

#if USE_TDM
  const unsigned ldsBase[2] = { (unsigned)(size_t)&sAB[0][0],
                                (unsigned)(size_t)&sAB[1][0] };
  auto issue = [&](int s) {
    int srcSel = s / kSteps;
    int k0 = (s - srcSel * kSteps) * 32;
    const __bf16* A = srcSel ? A1 : A0;
    int lda = srcSel ? lda1 : lda0;
    unsigned base = ldsBase[s & 1];
    tdm2d(A + (size_t)rowBase * lda + k0, base, 128, lda, M - rowBase);
    tdm2d(Bt + (size_t)colBase * ldb + srcSel * K + k0, base + 4096u * 2u, 64, ldb, 64);
  };
  if (wave == 0) issue(0);
#else
  auto loadRegs = [&](int s, uint4& ra0, uint4& ra1, uint4& rb) {
    int srcSel = s / kSteps;
    int k0 = (s - srcSel * kSteps) * 32;
    const __bf16* A = srcSel ? A1 : A0;
    int lda = srcSel ? lda1 : lda0;
    int r = tid >> 1, kp = (tid & 1) * 16;
    int grow = rowBase + r; if (grow > M - 1) grow = M - 1;   // clamp, no divergence
    const __bf16* ap = A + (size_t)grow * lda + k0 + kp;
    ra0 = *(const uint4*)(ap);
    ra1 = *(const uint4*)(ap + 8);
    int nn = tid >> 2, kq = (tid & 3) * 8;
    rb = *(const uint4*)(Bt + (size_t)(colBase + nn) * ldb + srcSel * K + k0 + kq);
  };
  auto storeRegs = [&](int buf, uint4 ra0, uint4 ra1, uint4 rb) {
    __bf16* sA = &sAB[buf][0];
    __bf16* sB = &sAB[buf][4096];
    int r = tid >> 1, kp = (tid & 1) * 16;
    *(uint4*)(sA + r * 32 + kp)     = ra0;
    *(uint4*)(sA + r * 32 + kp + 8) = ra1;
    int nn = tid >> 2, kq = (tid & 3) * 8;
    *(uint4*)(sB + nn * 32 + kq) = rb;
  };
  uint4 ra0, ra1, rb;
  loadRegs(0, ra0, ra1, rb);
#endif

  for (int i = 0; i < steps; ++i) {
    const int buf = i & 1;
#if USE_TDM
    if (wave == 0) __builtin_amdgcn_s_wait_tensorcnt(0);   // tile i landed
    __syncthreads();                                       // visible; prev buf free
    if (wave == 0 && i + 1 < steps) issue(i + 1);          // overlap DMA with WMMA
#else
    storeRegs(buf, ra0, ra1, rb);
    uint4 na0 = {}, na1 = {}, nb = {};
    if (i + 1 < steps) loadRegs(i + 1, na0, na1, nb);
    __syncthreads();
#endif
    const __bf16* sA = &sAB[buf][0];
    const __bf16* sB = &sAB[buf][4096];

    // fragments per ISA 7.12.2 lane layouts (contiguous ds_load_b128)
    FragBF a[2], b[2];
#pragma unroll
    for (int ii = 0; ii < 2; ++ii) {        // A 16x32: m=lane&15, kb=(lane>>4)*8
      int m  = wm * 32 + ii * 16 + (lane & 15);
      int kb = (lane >> 4) * 8;
      a[ii].q[0] = *(const uint4*)(sA + m * 32 + kb);
      a[ii].q[1] = *(const uint4*)(sA + m * 32 + kb + 16);
    }
#pragma unroll
    for (int jj = 0; jj < 2; ++jj) {        // B 32x16: n=lane&15, kb=(lane>>4)*8
      int n  = wn * 32 + jj * 16 + (lane & 15);
      int ks = (lane >> 4) * 8;
      b[jj].q[0] = *(const uint4*)(sB + n * 32 + ks);
      b[jj].q[1] = *(const uint4*)(sB + n * 32 + ks + 16);
    }
#pragma unroll
    for (int ii = 0; ii < 2; ++ii)
#pragma unroll
      for (int jj = 0; jj < 2; ++jj)
        acc[ii][jj] = __builtin_amdgcn_wmma_f32_16x16x32_bf16(
            false, a[ii].v, false, b[jj].v, (short)0, acc[ii][jj], false, false);
#if !USE_TDM
    ra0 = na0; ra1 = na1; rb = nb;
#endif
  }

  // ---- epilogue: bias + relu, bf16 C tile via LDS, coalesced b128 stores ----
  __syncthreads();                                  // done reading sAB buffers
  __bf16* sC = &sAB[0][0];                          // 128x64 bf16 = 16 KB
#pragma unroll
  for (int ii = 0; ii < 2; ++ii) {
    int mloc = wm * 32 + ii * 16 + (lane >> 4) * 8;
#pragma unroll
    for (int jj = 0; jj < 2; ++jj) {
      int nloc = wn * 32 + jj * 16 + (lane & 15);
      float bv = bias ? bias[colBase + nloc] : 0.0f;
#pragma unroll
      for (int r = 0; r < 8; ++r) {
        float v = acc[ii][jj][r] + bv;
        if (do_relu) v = fmaxf(v, 0.0f);
        sC[(mloc + r) * 64 + nloc] = (__bf16)v;
      }
    }
  }
  __syncthreads();
  {
    int m  = tid >> 1;                 // 0..127
    int n0 = (tid & 1) * 32;           // half row = 32 bf16 = 64 B
    int row = rowBase + m;
    if (row < M) {
      const uint4* s = (const uint4*)(sC + m * 64 + n0);
      uint4* d = (uint4*)(C + (size_t)row * ldc + colBase + n0);
#pragma unroll
      for (int j = 0; j < 4; ++j) d[j] = s[j];
    }
  }
}

// ---------------------------------------------------------------------------
// f32 -> bf16 bulk convert (8 elems / thread, b128 stores)
// ---------------------------------------------------------------------------
__global__ __launch_bounds__(256)
void k_cvt_bf16(__bf16* __restrict__ dst, const float* __restrict__ src, long long n8)
{
  long long i = (long long)blockIdx.x * 256 + threadIdx.x;
  if (i >= n8) return;
  const float4* s = (const float4*)(src + i * 8);
  float4 a = s[0], b = s[1];
  BF8 o;
  o.h[0] = (__bf16)a.x; o.h[1] = (__bf16)a.y; o.h[2] = (__bf16)a.z; o.h[3] = (__bf16)a.w;
  o.h[4] = (__bf16)b.x; o.h[5] = (__bf16)b.y; o.h[6] = (__bf16)b.z; o.h[7] = (__bf16)b.w;
  ((uint4*)dst)[i] = o.q;
}

// weight prep: transpose + convert  src f32 [K,Nc] -> dst bf16 [Nc,K]
__global__ void k_wprep(__bf16* __restrict__ dst, const float* __restrict__ src,
                        int K, int Nc)
{
  int i = blockIdx.x * 256 + threadIdx.x;
  if (i >= K * Nc) return;
  int n = i / K, k = i - n * K;
  dst[i] = (__bf16)src[(size_t)k * Nc + n];
}

__global__ void k_fill(float* __restrict__ p, float v, long long n) {
  long long i = (long long)blockIdx.x * blockDim.x + threadIdx.x;
  if (i < n) p[i] = v;
}

// ---------------------------------------------------------------------------
// Edge aggregation: agg_f32[dst] += h_bf16[src]  (one wave per edge)
// ---------------------------------------------------------------------------
__global__ __launch_bounds__(256)
void k_aggregate(const __bf16* __restrict__ h, int lda,
                 float* __restrict__ agg,
                 const int* __restrict__ src, const int* __restrict__ dst, int E)
{
  int e    = blockIdx.x * 8 + (threadIdx.x >> 5);
  int lane = threadIdx.x & 31;
  if (e >= E) return;
  int s = src[e], d = dst[e];
  const uint4* hp = (const uint4*)(h + (size_t)s * lda);   // 512 bf16 = 64 uint4
  float* ap = agg + (size_t)d * NH;
#pragma unroll
  for (int t = 0; t < 2; ++t) {
    int c = lane + 32 * t;           // uint4 index 0..63
    BF8 v; v.q = hp[c];
#pragma unroll
    for (int j = 0; j < 8; ++j)
      atomicAdd(ap + c * 8 + j, (float)v.h[j]);
  }
}

// ---------------------------------------------------------------------------
// Memory attention: h' = softmax(h @ mem^T) @ mem  (wave/node, mem in LDS)
// ---------------------------------------------------------------------------
__global__ __launch_bounds__(256)
void k_memattn(const __bf16* __restrict__ hin, int ldin,
               const float* __restrict__ mem,
               __bf16* __restrict__ hout, int ldout, int N)
{
  __shared__ float sm[NMEMS * NH];     // 16 KB
  for (int i = threadIdx.x; i < NMEMS * NH; i += 256) sm[i] = mem[i];
  __syncthreads();

  int node = blockIdx.x * 8 + (threadIdx.x >> 5);
  int lane = threadIdx.x & 31;
  if (node >= N) return;

  const __bf16* hp = hin + (size_t)node * ldin;
  float acc[NMEMS];
#pragma unroll
  for (int j = 0; j < NMEMS; ++j) acc[j] = 0.f;
  for (int k = lane; k < NH; k += 32) {
    float hv = (float)hp[k];
#pragma unroll
    for (int j = 0; j < NMEMS; ++j) acc[j] += hv * sm[j * NH + k];
  }
#pragma unroll
  for (int j = 0; j < NMEMS; ++j)
    for (int off = 16; off > 0; off >>= 1)
      acc[j] += __shfl_xor(acc[j], off, 32);

  float m = acc[0];
#pragma unroll
  for (int j = 1; j < NMEMS; ++j) m = fmaxf(m, acc[j]);
  float s = 0.f, p[NMEMS];
#pragma unroll
  for (int j = 0; j < NMEMS; ++j) { p[j] = __expf(acc[j] - m); s += p[j]; }
  float inv = 1.0f / s;

  __bf16* op = hout + (size_t)node * ldout;
  for (int c = lane; c < NH; c += 32) {
    float o = 0.f;
#pragma unroll
    for (int j = 0; j < NMEMS; ++j) o += p[j] * inv * sm[j * NH + c];
    op[c] = (__bf16)o;
  }
}

// ---------------------------------------------------------------------------
// Segment pooling (max + mean); ordered-uint atomicMax + f32 atomicAdd
// ---------------------------------------------------------------------------
__global__ void k_pool_init(unsigned* __restrict__ gmax, float* __restrict__ gsum,
                            float* __restrict__ gcnt) {
  int i = blockIdx.x * blockDim.x + threadIdx.x;
  if (i < NGRAPH * 2 * NH) { gmax[i] = encOrd(-3.0e38f); gsum[i] = 0.f; }
  if (i < NGRAPH) gcnt[i] = 0.f;
}

__global__ __launch_bounds__(256)
void k_pool(const __bf16* __restrict__ xs, const int* __restrict__ batch,
            unsigned* __restrict__ gmax, float* __restrict__ gsum,
            float* __restrict__ gcnt, int N)
{
  int node = blockIdx.x;
  if (node >= N) return;
  int g = batch[node];
  const __bf16* xp = xs + (size_t)node * (2 * NH);
  unsigned* gm = gmax + (size_t)g * (2 * NH);
  float*    gs = gsum + (size_t)g * (2 * NH);
  for (int f = threadIdx.x; f < 2 * NH; f += 256) {
    float v = (float)xp[f];
    atomicMax(gm + f, encOrd(v));
    atomicAdd(gs + f, v);
  }
  if (threadIdx.x == 0) atomicAdd(gcnt + g, 1.0f);
}

// ---------------------------------------------------------------------------
// Classifier: [G,2048]@[2048,10] + bias, log_softmax. One wave per graph.
// ---------------------------------------------------------------------------
__global__ __launch_bounds__(256)
void k_classify(const unsigned* __restrict__ gmax, const float* __restrict__ gsum,
                const float* __restrict__ gcnt,
                const float* __restrict__ lin_w, const float* __restrict__ lin_b,
                float* __restrict__ out)
{
  int g    = blockIdx.x * 8 + (threadIdx.x >> 5);
  int lane = threadIdx.x & 31;
  if (g >= NGRAPH) return;
  float inv = 1.0f / fmaxf(gcnt[g], 1.0f);

  float acc[NCLS];
#pragma unroll
  for (int c = 0; c < NCLS; ++c) acc[c] = 0.f;
  for (int k = lane; k < 4 * NH; k += 32) {
    float gv = (k < 2 * NH) ? decOrd(gmax[(size_t)g * 2 * NH + k])
                            : gsum[(size_t)g * 2 * NH + (k - 2 * NH)] * inv;
    const float* wr = lin_w + (size_t)k * NCLS;
#pragma unroll
    for (int c = 0; c < NCLS; ++c) acc[c] += gv * wr[c];
  }
#pragma unroll
  for (int c = 0; c < NCLS; ++c)
    for (int off = 16; off > 0; off >>= 1)
      acc[c] += __shfl_xor(acc[c], off, 32);

  if (lane == 0) {
    float logits[NCLS];
    float m = -3.0e38f;
#pragma unroll
    for (int c = 0; c < NCLS; ++c) { logits[c] = acc[c] + lin_b[c]; m = fmaxf(m, logits[c]); }
    float s = 0.f;
#pragma unroll
    for (int c = 0; c < NCLS; ++c) s += __expf(logits[c] - m);
    float lse = m + __logf(s);
#pragma unroll
    for (int c = 0; c < NCLS; ++c) out[(size_t)g * NCLS + c] = logits[c] - lse;
  }
}

// ---------------------------------------------------------------------------
extern "C" void kernel_launch(void* const* d_in, const int* in_sizes, int n_in,
                              void* d_out, int out_size, void* d_ws, size_t ws_size,
                              hipStream_t stream)
{
  const float* x     = (const float*)d_in[0];
  const int*   ei    = (const int*)d_in[1];
  const int*   batch = (const int*)d_in[2];
  const float* pre_w = (const float*)d_in[4];
  const float* pre_b = (const float*)d_in[5];
  const float* w1a[2]  = {(const float*)d_in[6],  (const float*)d_in[11]};
  const float* b1a[2]  = {(const float*)d_in[7],  (const float*)d_in[12]};
  const float* w2a[2]  = {(const float*)d_in[8],  (const float*)d_in[13]};
  const float* b2a[2]  = {(const float*)d_in[9],  (const float*)d_in[14]};
  const float* mema[2] = {(const float*)d_in[10], (const float*)d_in[15]};
  const float* lin_w = (const float*)d_in[16];
  const float* lin_b = (const float*)d_in[17];
  float* out = (float*)d_out;

  const int N = NNODES, E = NEDGES;

  // workspace carve-out (256B aligned chunks)
  char* w = (char*)d_ws;
  auto carve = [&](size_t bytes) { char* p = w; w += (bytes + 255) & ~(size_t)255; return p; };
  __bf16*   xb   = (__bf16*)  carve((size_t)N * NH * 2);       // x in bf16
  __bf16*   hb   = (__bf16*)  carve((size_t)N * NH * 2);       // pre_conv out
  float*    aggf = (float*)   carve((size_t)N * NH * 4);       // f32 scatter accum
  __bf16*   aggb = (__bf16*)  carve((size_t)N * NH * 2);       // agg bf16 / MLP2 out
  __bf16*   zb   = (__bf16*)  carve((size_t)N * NH * 2);       // MLP1 out
  __bf16*   xsb  = (__bf16*)  carve((size_t)N * 2 * NH * 2);   // layer outputs concat
  __bf16*   pwt  = (__bf16*)  carve((size_t)NH * NH * 2);
  __bf16*   w1t[2] = {(__bf16*)carve((size_t)NH * 2 * NH * 2),
                      (__bf16*)carve((size_t)NH * 2 * NH * 2)};
  __bf16*   w2t[2] = {(__bf16*)carve((size_t)NH * NH * 2),
                      (__bf16*)carve((size_t)NH * NH * 2)};
  unsigned* gmax = (unsigned*)carve((size_t)NGRAPH * 2 * NH * 4);
  float*    gsum = (float*)   carve((size_t)NGRAPH * 2 * NH * 4);
  float*    gcnt = (float*)   carve((size_t)NGRAPH * 4);

  const int* srcI = ei;
  const int* dstI = ei + E;

  dim3 blk(256);
  dim3 gGemm((N + 127) / 128, NH / 64);
  long long n8 = (long long)N * NH / 8;

  // one-time (per launch) precision/layout prep: activations bf16, weights bf16^T
  k_cvt_bf16<<<(unsigned)((n8 + 255) / 256), blk, 0, stream>>>(xb, x, n8);
  k_wprep<<<(NH * NH + 255) / 256, blk, 0, stream>>>(pwt, pre_w, NH, NH);
  for (int l = 0; l < 2; ++l) {
    k_wprep<<<(2 * NH * NH + 255) / 256, blk, 0, stream>>>(w1t[l], w1a[l], 2 * NH, NH);
    k_wprep<<<(NH * NH + 255) / 256, blk, 0, stream>>>(w2t[l], w2a[l], NH, NH);
  }

  // pre_conv: hb = x @ pre_w + pre_b
  k_gemm_wmma<<<gGemm, blk, 0, stream>>>(xb, NH, nullptr, 0, pwt, NH,
                                         pre_b, hb, NH, N, NH, 0);

  const __bf16* hcur = hb; int ldh = NH;
  for (int layer = 0; layer < 2; ++layer) {
    k_fill<<<(unsigned)(((size_t)N * NH + 255) / 256), blk, 0, stream>>>(
        aggf, 0.0f, (long long)N * NH);
    k_aggregate<<<(E + 7) / 8, blk, 0, stream>>>(hcur, ldh, aggf, srcI, dstI, E);
    k_cvt_bf16<<<(unsigned)((n8 + 255) / 256), blk, 0, stream>>>(aggb, aggf, n8);

    // zb = relu( [hcur || agg] @ w1 + b1 )  -- dual-source, one accumulator pass
    k_gemm_wmma<<<gGemm, blk, 0, stream>>>(hcur, ldh, aggb, NH,
                                           w1t[layer], 2 * NH,
                                           b1a[layer], zb, NH, N, NH, 1);
    // t = relu( zb @ w2 + b2 )   (aggb reused as t)
    k_gemm_wmma<<<gGemm, blk, 0, stream>>>(zb, NH, nullptr, 0, w2t[layer], NH,
                                           b2a[layer], aggb, NH, N, NH, 1);
    // memory attention -> xsb[:, layer*512 .. +512]
    k_memattn<<<(N + 7) / 8, blk, 0, stream>>>(aggb, NH, mema[layer],
                                               xsb + (size_t)layer * NH, 2 * NH, N);
    hcur = xsb + (size_t)layer * NH; ldh = 2 * NH;
  }

  k_pool_init<<<(NGRAPH * 2 * NH + 255) / 256, blk, 0, stream>>>(gmax, gsum, gcnt);
  k_pool<<<N, blk, 0, stream>>>(xsb, batch, gmax, gsum, gcnt, N);
  k_classify<<<NGRAPH / 8, blk, 0, stream>>>(gmax, gsum, gcnt, lin_w, lin_b, out);
}
